// WindowAttention_63299228009319
// MI455X (gfx1250) — compile-verified, matrix operand order checked by reading, not run on hardware
//
#include <hip/hip_runtime.h>
#include <hip/hip_bf16.h>

// ---------- CDNA5 WMMA types ----------
typedef __bf16 bf16;
typedef __attribute__((ext_vector_type(16))) __bf16 bf16x16;
typedef __attribute__((ext_vector_type(8)))  float  v8f;

#define WMMA_BF16(a, b, c) \
  __builtin_amdgcn_wmma_f32_16x16x32_bf16(false, (a), false, (b), (short)0, (c), false, false)

// Per-lane fragment layout for 16x16x32 bf16 WMMA (A 16x32 and B 32x16 share it):
// lanes 0-15  hold K = {0..7} U {16..23}; lanes 16-31 hold K = {8..15} U {24..31}
// -> per lane: two contiguous 16-byte chunks at p[k0] and p[k0+16], k0 = (lane>>4)*8.
__device__ __forceinline__ bf16x16 frag_from_bf16(const bf16* __restrict__ p, int k0) {
  bf16x16 r;
  uint4* rq = reinterpret_cast<uint4*>(&r);
  rq[0] = *reinterpret_cast<const uint4*>(p + k0);        // global_load_b128
  rq[1] = *reinterpret_cast<const uint4*>(p + k0 + 16);   // global_load_b128
  return r;
}

union Pack4 { bf16 h[4]; unsigned long long u; };

// ============================================================================
// Pre-convert kernels (run once per call; all bf16 from here on).
// ============================================================================

__global__ __launch_bounds__(256) void cvt_w_kernel(
    const float* __restrict__ src, bf16* __restrict__ dst, int n4)
{
  const int i = blockIdx.x * 256 + threadIdx.x;
  if (i < n4) {
    const float4 f = reinterpret_cast<const float4*>(src)[i];
    Pack4 o;
    o.h[0] = (bf16)f.x; o.h[1] = (bf16)f.y; o.h[2] = (bf16)f.z; o.h[3] = (bf16)f.w;
    *reinterpret_cast<unsigned long long*>(dst + (size_t)i * 4) = o.u;
  }
}

// x (fp32 [b][49][512]) -> xb (bf16 [b][64][512], rows 49..63 zeroed).
__global__ __launch_bounds__(128) void xpad_kernel(
    const float* __restrict__ x, bf16* __restrict__ xb)
{
  const int b   = blockIdx.x;
  const int row = blockIdx.y;                 // 0..63
  const int t   = threadIdx.x;                // 128 thr * 4 floats = 512 cols
  Pack4 o;
  if (row < 49) {
    const float4 f = *reinterpret_cast<const float4*>(
        x + ((size_t)b * 49 + row) * 512 + t * 4);
    o.h[0] = (bf16)f.x; o.h[1] = (bf16)f.y; o.h[2] = (bf16)f.z; o.h[3] = (bf16)f.w;
  } else {
    o.u = 0ull;
  }
  *reinterpret_cast<unsigned long long*>(
      xb + ((size_t)b * 64 + row) * 512 + t * 4) = o.u;
}

// ============================================================================
// Kernel 1: qkv = xb @ qkv_w^T + qkv_b.  Fully-unrolled K loop with a 1-stage
// software pipeline: loads for step k+1 (all immediate-offset b128) issue
// before the WMMAs of step k, so waits are partial and latency overlaps MMA.
// grid (2048, 12): blockIdx.y 0-3 -> q cols, 4-7 -> k cols, 8-11 -> v cols
// (scalar store branch). block 256 = 8 waves: 4 row-tiles x 2 col-halves.
// ============================================================================
__global__ __launch_bounds__(256) void qkv_kernel(
    const bf16* __restrict__ xb, const bf16* __restrict__ wb,
    const float* __restrict__ qkv_b,
    bf16* __restrict__ qb, bf16* __restrict__ kb, bf16* __restrict__ vTb)
{
  const int b      = blockIdx.x;
  const int wave   = threadIdx.x >> 5;
  const int lane   = threadIdx.x & 31;
  const int lane16 = lane & 15;
  const int hi     = lane >> 4;
  const int rt     = wave & 3;                 // row tile (16 tokens)
  const int cbase  = blockIdx.y * 128 + (wave >> 2) * 64;
  const int row    = rt * 16 + lane16;         // token row for A fragment
  const int k0     = hi * 8;
  const bf16* xrow = xb + ((size_t)b * 64 + row) * 512 + k0;  // pad rows zeroed
  const bf16* w0   = wb + (size_t)(cbase + lane16) * 512 + k0;

  v8f acc[4] = {};
  bf16x16 a_cur = frag_from_bf16(xrow, 0);
  bf16x16 b_cur[4];
#pragma unroll
  for (int t = 0; t < 4; ++t) b_cur[t] = frag_from_bf16(w0 + t * 16 * 512, 0);

#pragma unroll
  for (int kk = 0; kk < 512; kk += 32) {
    bf16x16 a_nxt;
    bf16x16 b_nxt[4];
    if (kk + 32 < 512) {                       // compile-time per unrolled step
      a_nxt = frag_from_bf16(xrow, kk + 32);
#pragma unroll
      for (int t = 0; t < 4; ++t)
        b_nxt[t] = frag_from_bf16(w0 + t * 16 * 512, kk + 32);
    }
#pragma unroll
    for (int t = 0; t < 4; ++t) acc[t] = WMMA_BF16(a_cur, b_cur[t], acc[t]);
    a_cur = a_nxt;                             // SSA-renamed, no real moves
#pragma unroll
    for (int t = 0; t < 4; ++t) b_cur[t] = b_nxt[t];
  }

  // ---- epilogue: scalar branch on q/k/v section ----
  const int which = blockIdx.y >> 2;           // uniform: 0=q 1=k 2=v
  const float qscale = 0.17677669529663687f;   // 32^-0.5
#pragma unroll
  for (int t = 0; t < 4; ++t) {
    const int c    = cbase + t * 16 + lane16;
    const float bs = qkv_b[c];
    const int cc = c & 511;
    const int h  = cc >> 5;
    const int d  = cc & 31;
    const size_t bh = (size_t)b * 16 + h;
    if (which == 0) {
#pragma unroll
      for (int r = 0; r < 8; ++r) {
        const int tok = rt * 16 + hi * 8 + r;  // C/D layout: M = r + hi*8
        qb[(bh * 64 + tok) * 32 + d] = (bf16)((acc[t][r] + bs) * qscale);
      }
    } else if (which == 1) {
#pragma unroll
      for (int r = 0; r < 8; ++r) {
        const int tok = rt * 16 + hi * 8 + r;
        kb[(bh * 64 + tok) * 32 + d] = (bf16)(acc[t][r] + bs);
      }
    } else {
#pragma unroll
      for (int r = 0; r < 8; ++r) {
        const int tok = rt * 16 + hi * 8 + r;
        vTb[(bh * 32 + d) * 64 + tok] = (bf16)(acc[t][r] + bs);  // transposed
      }
    }
  }
}

// ============================================================================
// Kernel 2: per (window, head): S = q@k^T ; softmax(S + relpos_bias + mask) ;
//           y = P @ V.  grid 2048*16, block 128 = 4 waves (one 16-row tile each).
// ============================================================================
__global__ __launch_bounds__(128) void attn_kernel(
    const bf16* __restrict__ qb, const bf16* __restrict__ kb,
    const bf16* __restrict__ vTb,
    const float* __restrict__ mask, const float* __restrict__ rel_tab,
    const int* __restrict__ rel_idx,
    bf16* __restrict__ y)
{
  __shared__ float S[64][65];   // stride 65 -> conflict-free row scans
  __shared__ bf16  P[64][72];   // stride 72 bf16 = 144 B, 16B-aligned frags

  const int bh     = blockIdx.x;
  const int b      = bh >> 4;
  const int h      = bh & 15;
  const int w      = b & 63;            // window index for mask
  const int tid    = threadIdx.x;
  const int rt     = tid >> 5;          // wave = row tile
  const int lane   = tid & 31;
  const int lane16 = lane & 15;
  const int hi     = lane >> 4;

  // ---- S = q @ k^T : one K=32 WMMA per 16x16 tile, 4 tiles per wave ----
  const bf16* qrow = qb + ((size_t)bh * 64 + rt * 16 + lane16) * 32;
  bf16x16 a = frag_from_bf16(qrow, hi * 8);
  bf16x16 km[4];
#pragma unroll
  for (int t = 0; t < 4; ++t)
    km[t] = frag_from_bf16(kb + ((size_t)bh * 64 + t * 16 + lane16) * 32, hi * 8);
  v8f sacc[4] = {};
#pragma unroll
  for (int t = 0; t < 4; ++t)
    sacc[t] = WMMA_BF16(a, km[t], sacc[t]);

#pragma unroll
  for (int t = 0; t < 4; ++t)
#pragma unroll
    for (int r = 0; r < 8; ++r)
      S[rt * 16 + hi * 8 + r][t * 16 + lane16] = sacc[t][r];
  __syncthreads();

  // ---- softmax rows with rel-pos bias + shifted-window mask ----
  if (tid < 64) {
    const int nq = tid;
    if (nq < 49) {
      const int*   ri   = rel_idx + nq * 49;
      const float* mrow = mask + ((size_t)w * 49 + nq) * 49;
      float mx = -1e30f;
      for (int m = 0; m < 49; ++m) {
        float s = S[nq][m] + rel_tab[ri[m] * 16 + h] + mrow[m];
        S[nq][m] = s;
        mx = fmaxf(mx, s);
      }
      float sum = 0.f;
      for (int m = 0; m < 49; ++m) {
        float e = __expf(S[nq][m] - mx);
        S[nq][m] = e;
        sum += e;
      }
      const float inv = 1.f / sum;
      for (int m = 0; m < 64; ++m)
        P[nq][m] = (m < 49) ? (bf16)(S[nq][m] * inv) : (bf16)0.f;
    } else {
      for (int m = 0; m < 64; ++m) P[nq][m] = (bf16)0.f;  // pad rows -> zeros
    }
  }
  __syncthreads();

  // ---- y = P @ V : K=64 over tokens (2 steps), 2 col tiles (hd=32) ----
  v8f oacc[2] = {};
  const bf16* prow = &P[rt * 16 + lane16][0];
#pragma unroll
  for (int kk = 0; kk < 64; kk += 32) {
    const int k0 = kk + hi * 8;
    bf16x16 a2 = frag_from_bf16(prow, k0);
#pragma unroll
    for (int ct = 0; ct < 2; ++ct) {
      bf16x16 b2 = frag_from_bf16(vTb + ((size_t)bh * 32 + ct * 16 + lane16) * 64, k0);
      oacc[ct] = WMMA_BF16(a2, b2, oacc[ct]);
    }
  }
#pragma unroll
  for (int ct = 0; ct < 2; ++ct)
#pragma unroll
    for (int r = 0; r < 8; ++r) {
      const int tok = rt * 16 + hi * 8 + r;
      const int c   = h * 32 + ct * 16 + lane16;
      y[((size_t)b * 64 + tok) * 512 + c] = (bf16)oacc[ct][r];
    }
}

// ============================================================================
// Kernel 3: out = y @ proj_w^T + proj_b (fp32 out, valid 49 rows only),
// same pipelined fully-unrolled K loop. grid (2048, 4), block 256 = 8 waves.
// ============================================================================
__global__ __launch_bounds__(256) void proj_kernel(
    const bf16* __restrict__ y, const bf16* __restrict__ wb,
    const float* __restrict__ proj_b, float* __restrict__ out)
{
  const int b      = blockIdx.x;
  const int wave   = threadIdx.x >> 5;
  const int lane   = threadIdx.x & 31;
  const int lane16 = lane & 15;
  const int hi     = lane >> 4;
  const int rt     = wave & 3;
  const int cbase  = blockIdx.y * 128 + (wave >> 2) * 64;
  const int row    = rt * 16 + lane16;
  const int k0     = hi * 8;
  const bf16* yrow = y + ((size_t)b * 64 + row) * 512 + k0;  // pad rows zeroed
  const bf16* w0   = wb + (size_t)(cbase + lane16) * 512 + k0;

  v8f acc[4] = {};
  bf16x16 a_cur = frag_from_bf16(yrow, 0);
  bf16x16 b_cur[4];
#pragma unroll
  for (int t = 0; t < 4; ++t) b_cur[t] = frag_from_bf16(w0 + t * 16 * 512, 0);

#pragma unroll
  for (int kk = 0; kk < 512; kk += 32) {
    bf16x16 a_nxt;
    bf16x16 b_nxt[4];
    if (kk + 32 < 512) {
      a_nxt = frag_from_bf16(yrow, kk + 32);
#pragma unroll
      for (int t = 0; t < 4; ++t)
        b_nxt[t] = frag_from_bf16(w0 + t * 16 * 512, kk + 32);
    }
#pragma unroll
    for (int t = 0; t < 4; ++t) acc[t] = WMMA_BF16(a_cur, b_cur[t], acc[t]);
    a_cur = a_nxt;
#pragma unroll
    for (int t = 0; t < 4; ++t) b_cur[t] = b_nxt[t];
  }

#pragma unroll
  for (int t = 0; t < 4; ++t) {
    const int c    = cbase + t * 16 + lane16;
    const float bs = proj_b[c];
#pragma unroll
    for (int r = 0; r < 8; ++r) {
      const int tok = rt * 16 + hi * 8 + r;
      if (tok < 49)
        out[((size_t)b * 49 + tok) * 512 + c] = acc[t][r] + bs;
    }
  }
}

// ============================================================================
extern "C" void kernel_launch(void* const* d_in, const int* in_sizes, int n_in,
                              void* d_out, int out_size, void* d_ws, size_t ws_size,
                              hipStream_t stream) {
  (void)in_sizes; (void)n_in; (void)out_size; (void)ws_size;
  const float* x       = (const float*)d_in[0];
  const float* mask    = (const float*)d_in[1];
  const float* qkv_w   = (const float*)d_in[2];
  const float* qkv_b   = (const float*)d_in[3];
  const float* proj_w  = (const float*)d_in[4];
  const float* proj_b  = (const float*)d_in[5];
  const float* rel_tab = (const float*)d_in[6];
  const int*   rel_idx = (const int*)d_in[7];
  float* out = (float*)d_out;

  // workspace slabs: q, k ([bh][64][32] bf16), vT ([bh][32][64]), y ([b][64][512]),
  // xb ([b][64][512]) -- 128 MB each; then bf16 weight copies.
  const size_t SZ = (size_t)2048 * 16 * 64 * 32 * sizeof(bf16);
  bf16* qb   = (bf16*)d_ws;
  bf16* kb   = (bf16*)((char*)d_ws + SZ);
  bf16* vTb  = (bf16*)((char*)d_ws + 2 * SZ);
  bf16* yb   = (bf16*)((char*)d_ws + 3 * SZ);
  bf16* xbuf = (bf16*)((char*)d_ws + 4 * SZ);
  bf16* qwb  = (bf16*)((char*)d_ws + 5 * SZ);
  bf16* pwb  = qwb + (size_t)1536 * 512;

  const int n4_qw = 1536 * 512 / 4;   // 196608
  const int n4_pw = 512 * 512 / 4;    //  65536
  cvt_w_kernel<<<(n4_qw + 255) / 256, 256, 0, stream>>>(qkv_w, qwb, n4_qw);
  cvt_w_kernel<<<(n4_pw + 255) / 256, 256, 0, stream>>>(proj_w, pwb, n4_pw);
  xpad_kernel<<<dim3(2048, 64), 128, 0, stream>>>(x, xbuf);

  qkv_kernel<<<dim3(2048, 12), 256, 0, stream>>>(xbuf, qwb, qkv_b, qb, kb, vTb);
  attn_kernel<<<dim3(2048 * 16), 128, 0, stream>>>(qb, kb, vTb, mask, rel_tab, rel_idx, yb);
  proj_kernel<<<dim3(2048, 4), 256, 0, stream>>>(yb, pwb, proj_b, out);
}
